// SRTransformer_80513456931534
// MI455X (gfx1250) — matrix operand order repaired and checked by reading, not compiled, not run on hardware
//
#include <hip/hip_runtime.h>
#include <hip/hip_bf16.h>
#include <stdint.h>

// Problem constants (fixed by the reference).
#define WB 2
#define WN 8192
#define WF 64
#define WH 64
#define WK 16
#define NPTS (WB * WN) // 16384

typedef __bf16 bf16;
typedef __bf16 v16bf __attribute__((ext_vector_type(16)));
typedef __bf16 v8bf  __attribute__((ext_vector_type(8)));
typedef float  v8f   __attribute__((ext_vector_type(8)));
typedef unsigned int v4u __attribute__((ext_vector_type(4)));
typedef int v8i __attribute__((ext_vector_type(8)));
typedef int v4i __attribute__((ext_vector_type(4)));

#if __has_builtin(__builtin_amdgcn_tensor_load_to_lds) && __has_builtin(__builtin_amdgcn_s_wait_tensorcnt)
#define HAVE_TDM 1
#else
#define HAVE_TDM 0
#endif

// ---------------------------------------------------------------------------
// WMMA helpers (wave32, V_WMMA_F32_16X16X32_BF16)
// ---------------------------------------------------------------------------
__device__ __forceinline__ v8f wmma_bf16(v16bf a, v16bf b, v8f c) {
  return __builtin_amdgcn_wmma_f32_16x16x32_bf16(false, a, false, b,
                                                 (short)0, c, false, false);
}

// A fragment (16x32 bf16) from a row-major f32 matrix M[16 x ld], cols k0..k0+31.
// ISA 7.12.2: lane holds row m = lane&15; its 16 elements are two contiguous
// 8-wide K runs: [k0+kh .. +7] and [k0+16+kh .. +7], kh = 8*(lane>=16).
// kmax must be a multiple of 8 aligned with the runs (true for 16/64 here).
__device__ __forceinline__ v16bf load_A_f32(const float* M, int ld, int k0, int kmax) {
  int lane = threadIdx.x & 31;
  int m = lane & 15;
  int kh = (lane >> 4) * 8;
  const float* row = M + m * ld;
  int kb0 = k0 + kh;
  int kb1 = k0 + 16 + kh;
  float4 z = {0.f, 0.f, 0.f, 0.f};
  float4 r0a = z, r0b = z, r1a = z, r1b = z;
  if (kb0 < kmax) {
    r0a = *(const float4*)(row + kb0);
    r0b = *(const float4*)(row + kb0 + 4);
  }
  if (kb1 < kmax) {
    r1a = *(const float4*)(row + kb1);
    r1b = *(const float4*)(row + kb1 + 4);
  }
  v16bf a;
  a[0] = (bf16)r0a.x;  a[1] = (bf16)r0a.y;  a[2] = (bf16)r0a.z;  a[3] = (bf16)r0a.w;
  a[4] = (bf16)r0b.x;  a[5] = (bf16)r0b.y;  a[6] = (bf16)r0b.z;  a[7] = (bf16)r0b.w;
  a[8] = (bf16)r1a.x;  a[9] = (bf16)r1a.y;  a[10] = (bf16)r1a.z; a[11] = (bf16)r1a.w;
  a[12] = (bf16)r1b.x; a[13] = (bf16)r1b.y; a[14] = (bf16)r1b.z; a[15] = (bf16)r1b.w;
  return a;
}

// B fragment (32x16 bf16) from a PRE-SWIZZLED fragment buffer: each fragment is
// 32 lanes x 16 contiguous bf16 (1 KB), so a load is two ds_load_b128 per lane.
__device__ __forceinline__ v16bf load_B_frag(const bf16* base) {
  int lane = threadIdx.x & 31;
  const v8bf* p = (const v8bf*)(base + lane * 16);
  v8bf lo = p[0];
  v8bf hi = p[1];
  v16bf b;
#pragma unroll
  for (int e = 0; e < 8; ++e) { b[e] = lo[e]; b[e + 8] = hi[e]; }
  return b;
}

// Store C/D tile (16x16 f32) into row-major M[16 x ld] at col n0, adding bias,
// optional ReLU. Lane holds N=lane&15, rows r + 8*(lane>>4).
__device__ __forceinline__ void store_C(float* M, int ld, int n0, v8f c,
                                        const float* bias, bool relu) {
  int lane = threadIdx.x & 31;
  int n = n0 + (lane & 15);
  int mb = (lane >> 4) << 3;
  float bv = bias ? bias[n] : 0.0f;
#pragma unroll
  for (int r = 0; r < 8; ++r) {
    float v = c[r] + bv;
    if (relu) v = fmaxf(v, 0.0f);
    M[(mb + r) * ld + n] = v;
  }
}

// ---------------------------------------------------------------------------
// Tensor Data Mover: 1-D tile load global -> LDS (bf16 weights).
// D# packing per ISA 08_async_tensor §8.3/8.4 (group0 / group1).
// ---------------------------------------------------------------------------
__device__ __forceinline__ uint32_t lds_off_of(const void* p) {
  // Flat LDS addresses carry the aperture in bits [63:32]; low 32 bits are the
  // LDS byte offset (ISA 10.2 aperture calculation).
  return (uint32_t)(uint64_t)(uintptr_t)p;
}

#if HAVE_TDM
__device__ __forceinline__ void tdm_load_1d_bf16(uint32_t lds_byte_off, uint64_t ga,
                                                 uint32_t elems) {
  v4u g0;
  g0[0] = 1u;                                             // count=1, user desc
  g0[1] = lds_byte_off;                                   // lds_addr
  g0[2] = (uint32_t)(ga & 0xFFFFFFFFull);                 // global_addr[31:0]
  g0[3] = (uint32_t)((ga >> 32) & 0x1FFFFFFull) | (2u << 30); // addr[56:32] | type=2
  v8i g1;
  g1[0] = (int)(1u << 16);                                // data_size=1 (2 bytes)
  g1[1] = (int)((elems & 0xFFFFu) << 16);                 // tensor_dim0[15:0]
  g1[2] = (int)(((elems >> 16) & 0xFFFFu) | (1u << 16));  // dim0 hi | tensor_dim1=1
  g1[3] = (int)((elems & 0xFFFFu) << 16);                 // tile_dim0 = elems
  g1[4] = 0;                                              // tile_dim1/2 unused (1-D)
  g1[5] = (int)elems;                                     // tensor_dim0_stride
  g1[6] = 0;
  g1[7] = 0;
  v4i z4 = {0, 0, 0, 0};
  v8i z8 = {0, 0, 0, 0, 0, 0, 0, 0};
  // clang-23 / therock-10.0 arity: (g0, g1, g2, g3, g4, cpol)
  __builtin_amdgcn_tensor_load_to_lds(g0, g1, z4, z4, z8, 0);
}
#endif

// Stage `elems` bf16 weights from global ws into LDS (block-cooperative).
// Caller must __syncthreads() afterwards.
__device__ __forceinline__ void stage_weights(bf16* dst, const unsigned short* src,
                                              int elems) {
#if HAVE_TDM
  if (threadIdx.x < 32) {
    tdm_load_1d_bf16(lds_off_of(dst), (uint64_t)(uintptr_t)src, (uint32_t)elems);
    __builtin_amdgcn_s_wait_tensorcnt(0);
  }
#else
  for (int i = threadIdx.x; i < elems; i += blockDim.x) {
    unsigned short u = src[i];
    dst[i] = *reinterpret_cast<bf16*>(&u);
  }
#endif
}

// ---------------------------------------------------------------------------
// Kernel W: swizzle f32 weights [K x N] into WMMA B-fragment-ordered bf16.
// Fragment (kt,nt): lane L element e holds W[kt*32 + 16*(e>=8) + 8*(L>=16) + (e&7)][nt*16 + (L&15)],
// stored at dst[frag*512 + L*16 + e]. K is zero-padded up to a multiple of 32.
// ---------------------------------------------------------------------------
__global__ void swizzle_w_kernel(const float* __restrict__ src,
                                 unsigned short* __restrict__ dst, int K, int N) {
  int Kpad = (K + 31) & ~31;
  int total = Kpad * N;
  int i = blockIdx.x * 256 + threadIdx.x;
  if (i >= total) return;
  int frag = i >> 9;
  int lane = (i >> 4) & 31;
  int e = i & 15;
  int NT = N >> 4;
  int kt = frag / NT, nt = frag % NT;
  int k = kt * 32 + ((e >> 3) << 4) + ((lane >> 4) << 3) + (e & 7);
  int n = nt * 16 + (lane & 15);
  float v = (k < K) ? src[k * N + n] : 0.0f;
  bf16 h = (bf16)v;
  dst[i] = *reinterpret_cast<unsigned short*>(&h);
}

// ---------------------------------------------------------------------------
// Kernel A: exact KNN (top-16 by squared distance, ascending, ties -> lower idx).
// One query per thread; candidates tiled through LDS.
// ---------------------------------------------------------------------------
__global__ void __launch_bounds__(256)
knn_kernel(const float* __restrict__ xyzp, int* __restrict__ knn_idx) {
  int gq = blockIdx.x * 256 + threadIdx.x;  // block spans one batch's points
  int b = gq / WN;
  const float* p = xyzp + (size_t)gq * 4;
  float qx = p[0], qy = p[1], qz = p[2];

  float bd[WK];
  int bi[WK];
#pragma unroll
  for (int j = 0; j < WK; ++j) { bd[j] = 3.4e38f; bi[j] = 0; }

  __shared__ float sp[256 * 3];
  const float* base = xyzp + (size_t)b * WN * 4;
  for (int t0 = 0; t0 < WN; t0 += 256) {
    __syncthreads();
    {
      const float* c = base + (size_t)(t0 + threadIdx.x) * 4;
      sp[threadIdx.x * 3 + 0] = c[0];
      sp[threadIdx.x * 3 + 1] = c[1];
      sp[threadIdx.x * 3 + 2] = c[2];
    }
    __syncthreads();
    for (int j = 0; j < 256; ++j) {
      float dx = qx - sp[j * 3 + 0];
      float dy = qy - sp[j * 3 + 1];
      float dz = qz - sp[j * 3 + 2];
      float d = dx * dx + dy * dy + dz * dz;
      if (d < bd[WK - 1]) {                 // rare: bubble-insert (static indices)
        bd[WK - 1] = d;
        bi[WK - 1] = t0 + j;
#pragma unroll
        for (int s = WK - 1; s >= 1; --s) {
          if (bd[s] < bd[s - 1]) {
            float td = bd[s]; bd[s] = bd[s - 1]; bd[s - 1] = td;
            int ti = bi[s]; bi[s] = bi[s - 1]; bi[s - 1] = ti;
          }
        }
      }
    }
  }
  int* o = knn_idx + (size_t)gq * WK;
#pragma unroll
  for (int j = 0; j < WK; ++j) o[j] = bi[j];
}

// ---------------------------------------------------------------------------
// Kernel B: x = feat@Wker+b; q = x@Wq; k = x@Wk; v = x@Wv   (WMMA bf16)
// 128 threads = 4 waves, each wave owns a 16-row tile. Weights are
// fragment-ordered: 8 frags (2 kt x 4 nt) of 512 bf16 per 64x64 matrix.
// ---------------------------------------------------------------------------
__global__ void __launch_bounds__(128)
feat_xform_kernel(const float* __restrict__ feat,
                  const unsigned short* __restrict__ w4,  // ker|q|k|v frag-bf16
                  const float* __restrict__ bker,
                  float* __restrict__ qb, float* __restrict__ kb,
                  float* __restrict__ vb) {
  __shared__ __align__(16) bf16 sW[4 * 4096];       // 32 KB
  __shared__ float sbias[64];
  __shared__ __align__(16) float sx[4][16 * WH];    // 16 KB: per-wave x staging

  stage_weights(sW, w4, 4 * 4096);
  if (threadIdx.x < 64) sbias[threadIdx.x] = bker[threadIdx.x];
  __syncthreads();

  const bf16* Wker = sW;
  const bf16* Wq = sW + 4096;
  const bf16* Wk = sW + 8192;
  const bf16* Wv = sW + 12288;

  int wave = threadIdx.x >> 5;
  int tile = blockIdx.x * 4 + wave;
  const float* A = feat + (size_t)tile * 16 * WF;
  float* xs = sx[wave];

  v16bf a0 = load_A_f32(A, WF, 0, WF);
  v16bf a1 = load_A_f32(A, WF, 32, WF);
#pragma unroll
  for (int nt = 0; nt < 4; ++nt) {
    v8f c = {};
    c = wmma_bf16(a0, load_B_frag(Wker + (0 * 4 + nt) * 512), c);
    c = wmma_bf16(a1, load_B_frag(Wker + (1 * 4 + nt) * 512), c);
    store_C(xs, WH, nt * 16, c, sbias, false);
  }
  __syncthreads();

  v16bf x0 = load_A_f32(xs, WH, 0, WH);
  v16bf x1 = load_A_f32(xs, WH, 32, WH);
  float* outs[3] = {qb, kb, vb};
  const bf16* ws3[3] = {Wq, Wk, Wv};
#pragma unroll
  for (int m = 0; m < 3; ++m) {
    float* o = outs[m] + (size_t)tile * 16 * WH;
#pragma unroll
    for (int nt = 0; nt < 4; ++nt) {
      v8f c = {};
      c = wmma_bf16(x0, load_B_frag(ws3[m] + (0 * 4 + nt) * 512), c);
      c = wmma_bf16(x1, load_B_frag(ws3[m] + (1 * 4 + nt) * 512), c);
      store_C(o, WH, nt * 16, c, nullptr, false);
    }
  }
}

// ---------------------------------------------------------------------------
// Kernel C: per-point attention. One wave per point; K=16 neighbors.
//   pos_enc = relu(pos_cat@Wp1+b)@Wp2+b            (WMMA, K padded to 32)
//   a       = q - k + pos_enc
//   a2      = relu(a@Wa1+b)@Wa2+b                  (WMMA)
//   res     = sum_k softmax_k(a2/8) * (v + pos_enc)
// Fragment-ordered weights: Wp1 (Kpad 32): 4 frags; others 8 frags each.
// ---------------------------------------------------------------------------
__global__ void __launch_bounds__(64)
point_attn_kernel(const float* __restrict__ xyzp, const int* __restrict__ knn_idx,
                  const unsigned short* __restrict__ wmlp,  // p1|p2|a1|a2 frag-bf16
                  const float* __restrict__ bp1, const float* __restrict__ bp2,
                  const float* __restrict__ ba1, const float* __restrict__ ba2,
                  const float* __restrict__ qb, const float* __restrict__ kb,
                  const float* __restrict__ vb, float* __restrict__ resb) {
  __shared__ __align__(16) bf16 sW[14336];   // 28 KB: p1(2048)|p2|a1|a2(4096 ea)
  __shared__ float sb1[64], sb2[64], sb3[64], sb4[64];
  __shared__ __align__(16) float S0[2][16 * WH], S1[2][16 * WH], S2[2][16 * WH];
  __shared__ __align__(16) float sposcat[2][16 * 16];

  stage_weights(sW, wmlp, 14336);
  if (threadIdx.x < 64) {
    sb1[threadIdx.x] = bp1[threadIdx.x];
    sb2[threadIdx.x] = bp2[threadIdx.x];
    sb3[threadIdx.x] = ba1[threadIdx.x];
    sb4[threadIdx.x] = ba2[threadIdx.x];
  }
  __syncthreads();

  const bf16* Wp1 = sW;
  const bf16* Wp2 = sW + 2048;
  const bf16* Wa1 = sW + 6144;
  const bf16* Wa2 = sW + 10240;

  int wave = threadIdx.x >> 5;
  int lane = threadIdx.x & 31;
  int pt = blockIdx.x * 2 + wave;
  int b = pt / WN;
  const int* nidx = knn_idx + (size_t)pt * WK;

  float* pc = sposcat[wave];
  float* s0 = S0[wave];
  float* s1 = S1[wave];
  float* s2 = S2[wave];

  // Build pos_cat: row k = [pos(4) | pos_k(4) | diff(4) | |diff|(4)]
  if (lane < 16) {
    const float* pp = xyzp + (size_t)pt * 4;
    float px = pp[0], py = pp[1], pz = pp[2], pw = pp[3];
    int j = nidx[lane];
    const float* pk = xyzp + ((size_t)b * WN + j) * 4;
    float kx = pk[0], ky = pk[1], kz = pk[2], kw = pk[3];
    float dx = px - kx, dy = py - ky, dz = pz - kz, dw = pw - kw;
    float* r = pc + lane * 16;
    r[0] = px;  r[1] = py;  r[2] = pz;  r[3] = pw;
    r[4] = kx;  r[5] = ky;  r[6] = kz;  r[7] = kw;
    r[8] = dx;  r[9] = dy;  r[10] = dz; r[11] = dw;
    r[12] = fabsf(dx); r[13] = fabsf(dy); r[14] = fabsf(dz); r[15] = fabsf(dw);
  }
  __syncthreads();

  // h = relu(pos_cat @ Wp1 + bp1) -> s0   (K=16, padded to 32 on both sides)
  {
    v16bf pa = load_A_f32(pc, 16, 0, 16);
#pragma unroll
    for (int nt = 0; nt < 4; ++nt) {
      v8f c = {};
      c = wmma_bf16(pa, load_B_frag(Wp1 + nt * 512), c);
      store_C(s0, WH, nt * 16, c, sb1, true);
    }
  }
  __syncthreads();

  // pos_enc = h @ Wp2 + bp2 -> s1
  {
    v16bf h0 = load_A_f32(s0, WH, 0, WH);
    v16bf h1 = load_A_f32(s0, WH, 32, WH);
#pragma unroll
    for (int nt = 0; nt < 4; ++nt) {
      v8f c = {};
      c = wmma_bf16(h0, load_B_frag(Wp2 + (0 * 4 + nt) * 512), c);
      c = wmma_bf16(h1, load_B_frag(Wp2 + (1 * 4 + nt) * 512), c);
      store_C(s1, WH, nt * 16, c, sb2, false);
    }
  }
  __syncthreads();

  // a = q - k_gathered + pos_enc -> s0  (each lane owns a half-row of 32 floats)
  {
    const float* qrow = qb + (size_t)pt * WH;
    int row = lane >> 1;
    int cb = (lane & 1) * 32;
    int j = nidx[row];
    const float* krow = kb + ((size_t)b * WN + j) * WH;
#pragma unroll
    for (int t = 0; t < 8; ++t) {
      int c0 = cb + t * 4;
      float4 qv = *(const float4*)(qrow + c0);
      float4 kv = *(const float4*)(krow + c0);
      float4 pe = *(const float4*)(s1 + row * WH + c0);
      float4 o;
      o.x = qv.x - kv.x + pe.x;
      o.y = qv.y - kv.y + pe.y;
      o.z = qv.z - kv.z + pe.z;
      o.w = qv.w - kv.w + pe.w;
      *(float4*)(s0 + row * WH + c0) = o;
    }
  }
  __syncthreads();

  // t = relu(a @ Wa1 + ba1) -> s2
  {
    v16bf a0 = load_A_f32(s0, WH, 0, WH);
    v16bf a1 = load_A_f32(s0, WH, 32, WH);
#pragma unroll
    for (int nt = 0; nt < 4; ++nt) {
      v8f c = {};
      c = wmma_bf16(a0, load_B_frag(Wa1 + (0 * 4 + nt) * 512), c);
      c = wmma_bf16(a1, load_B_frag(Wa1 + (1 * 4 + nt) * 512), c);
      store_C(s2, WH, nt * 16, c, sb3, true);
    }
  }
  __syncthreads();

  // a2 = t @ Wa2 + ba2 -> s0
  {
    v16bf t0 = load_A_f32(s2, WH, 0, WH);
    v16bf t1 = load_A_f32(s2, WH, 32, WH);
#pragma unroll
    for (int nt = 0; nt < 4; ++nt) {
      v8f c = {};
      c = wmma_bf16(t0, load_B_frag(Wa2 + (0 * 4 + nt) * 512), c);
      c = wmma_bf16(t1, load_B_frag(Wa2 + (1 * 4 + nt) * 512), c);
      store_C(s0, WH, nt * 16, c, sb4, false);
    }
  }
  __syncthreads();

  // softmax over k (rows) per column, then weighted sum with (v + pos_enc).
  {
    const float scale = 0.125f;  // 1/sqrt(64)
#pragma unroll
    for (int cc = 0; cc < 2; ++cc) {
      int c = lane + cc * 32;
      float m = -3.4e38f;
#pragma unroll
      for (int k = 0; k < WK; ++k) m = fmaxf(m, s0[k * WH + c] * scale);
      float ex[WK];
      float sum = 0.0f;
#pragma unroll
      for (int k = 0; k < WK; ++k) {
        ex[k] = __expf(s0[k * WH + c] * scale - m);
        sum += ex[k];
      }
      float acc = 0.0f;
#pragma unroll
      for (int k = 0; k < WK; ++k) {
        int j = nidx[k];
        acc += ex[k] * (vb[((size_t)b * WN + j) * WH + c] + s1[k * WH + c]);
      }
      resb[(size_t)pt * WH + c] = acc / sum;
    }
  }
}

// ---------------------------------------------------------------------------
// Kernel D: out = res @ Wagg + bagg + features   (WMMA)
// ---------------------------------------------------------------------------
__global__ void __launch_bounds__(128)
aggregate_kernel(const float* __restrict__ resb, const float* __restrict__ feat,
                 const unsigned short* __restrict__ wagg,
                 const float* __restrict__ bagg, float* __restrict__ out) {
  __shared__ __align__(16) bf16 sW[4096];
  __shared__ float sb[64];
  stage_weights(sW, wagg, 4096);
  if (threadIdx.x < 64) sb[threadIdx.x] = bagg[threadIdx.x];
  __syncthreads();

  int wave = threadIdx.x >> 5;
  int lane = threadIdx.x & 31;
  int tile = blockIdx.x * 4 + wave;
  const float* A = resb + (size_t)tile * 16 * WH;
  v16bf a0 = load_A_f32(A, WH, 0, WH);
  v16bf a1 = load_A_f32(A, WH, 32, WH);
#pragma unroll
  for (int nt = 0; nt < 4; ++nt) {
    v8f c = {};
    c = wmma_bf16(a0, load_B_frag(sW + (0 * 4 + nt) * 512), c);
    c = wmma_bf16(a1, load_B_frag(sW + (1 * 4 + nt) * 512), c);
    int nn = nt * 16 + (lane & 15);
    int mb = (lane >> 4) << 3;
#pragma unroll
    for (int r = 0; r < 8; ++r) {
      size_t row = (size_t)tile * 16 + mb + r;
      out[row * WF + nn] = c[r] + sb[nn] + feat[row * WF + nn];
    }
  }
}

// ---------------------------------------------------------------------------
// Host launch
// ---------------------------------------------------------------------------
extern "C" void kernel_launch(void* const* d_in, const int* in_sizes, int n_in,
                              void* d_out, int out_size, void* d_ws, size_t ws_size,
                              hipStream_t stream) {
  (void)in_sizes; (void)n_in; (void)out_size; (void)ws_size;

  const float* xyzp = (const float*)d_in[0];
  const float* feat = (const float*)d_in[1];
  const float* Wker = (const float*)d_in[2];
  const float* bker = (const float*)d_in[3];
  const float* Wq   = (const float*)d_in[4];
  const float* Wk   = (const float*)d_in[5];
  const float* Wv   = (const float*)d_in[6];
  const float* Wp1  = (const float*)d_in[7];
  const float* bp1  = (const float*)d_in[8];
  const float* Wp2  = (const float*)d_in[9];
  const float* bp2  = (const float*)d_in[10];
  const float* Wa1  = (const float*)d_in[11];
  const float* ba1  = (const float*)d_in[12];
  const float* Wa2  = (const float*)d_in[13];
  const float* ba2  = (const float*)d_in[14];
  const float* Wagg = (const float*)d_in[15];
  const float* bagg = (const float*)d_in[16];

  // Workspace layout.
  char* ws = (char*)d_ws;
  int* knn_idx = (int*)ws;                                  // NPTS*16 ints (1 MB)
  float* qb    = (float*)(ws + ((size_t)NPTS * WK * 4));
  float* kb    = qb + (size_t)NPTS * WH;
  float* vb    = kb + (size_t)NPTS * WH;
  float* resb  = vb + (size_t)NPTS * WH;
  unsigned short* wbf = (unsigned short*)(resb + (size_t)NPTS * WH);
  // Fragment-ordered bf16 weights (elements):
  //   block0: ker|q|k|v           (4 x 4096 = 16384)
  //   block1: p1(pad 2048)|p2|a1|a2 (2048 + 3 x 4096 = 14336)
  //   block2: agg                 (4096)
  unsigned short* w_ker = wbf + 0;
  unsigned short* w_q   = wbf + 4096;
  unsigned short* w_k   = wbf + 8192;
  unsigned short* w_v   = wbf + 12288;
  unsigned short* w_p1  = wbf + 16384;
  unsigned short* w_p2  = wbf + 18432;
  unsigned short* w_a1  = wbf + 22528;
  unsigned short* w_a2  = wbf + 26624;
  unsigned short* w_agg = wbf + 30720;

  auto swz = [&](const float* s, unsigned short* d, int K, int N) {
    int kpad = (K + 31) & ~31;
    int total = kpad * N;
    swizzle_w_kernel<<<(total + 255) / 256, 256, 0, stream>>>(s, d, K, N);
  };
  swz(Wker, w_ker, 64, 64);
  swz(Wq,   w_q,   64, 64);
  swz(Wk,   w_k,   64, 64);
  swz(Wv,   w_v,   64, 64);
  swz(Wp1,  w_p1,  16, 64);   // K padded to 32 with zeros
  swz(Wp2,  w_p2,  64, 64);
  swz(Wa1,  w_a1,  64, 64);
  swz(Wa2,  w_a2,  64, 64);
  swz(Wagg, w_agg, 64, 64);

  knn_kernel<<<NPTS / 256, 256, 0, stream>>>(xyzp, knn_idx);

  feat_xform_kernel<<<NPTS / 64, 128, 0, stream>>>(feat, w_ker, bker, qb, kb, vb);

  point_attn_kernel<<<NPTS / 2, 64, 0, stream>>>(xyzp, knn_idx, w_p1,
                                                 bp1, bp2, ba1, ba2,
                                                 qb, kb, vb, resb);

  aggregate_kernel<<<NPTS / 64, 128, 0, stream>>>(resb, feat, w_agg, bagg,
                                                  (float*)d_out);
}
